// Attention_8753143349560
// MI455X (gfx1250) — matrix operand order repaired and checked by reading, not compiled, not run on hardware
//
#include <hip/hip_runtime.h>

typedef __attribute__((ext_vector_type(16))) __bf16 v16bf;
typedef __attribute__((ext_vector_type(8)))  __bf16 v8bf;
typedef __attribute__((ext_vector_type(8)))  float  v8f;

#define B_   32
#define N_   577
#define C_   768
#define H_   12
#define HD_  64
#define M_   (B_ * N_)        // 18464 = 577 * 32
#define QKVN_ 2304
#define VSTRIDE_ 608          // padded V^T row stride (>= 576 + 32 read by last key tile, mult of 16)
#define SCALE_ 0.125f         // 64^-0.5

// GEMM block tiling
#define TM_  128
#define TN_  128
#define TK_  32
#define LDK_ 40               // padded LDS K-stride (elems): 80B -> 16-lane b128 reads
                              // start at dwords 20*l mod 64 = all-distinct -> bank-conflict-free

union V16U { v16bf v; v8bf h[2]; };

__device__ __forceinline__ int imin_(int a, int b) { return a < b ? a : b; }

__device__ __forceinline__ float rowmax16(float v) {
  v = fmaxf(v, __shfl_xor(v, 1));
  v = fmaxf(v, __shfl_xor(v, 2));
  v = fmaxf(v, __shfl_xor(v, 4));
  v = fmaxf(v, __shfl_xor(v, 8));
  return v;
}
__device__ __forceinline__ float rowsum16(float v) {
  v += __shfl_xor(v, 1);
  v += __shfl_xor(v, 2);
  v += __shfl_xor(v, 4);
  v += __shfl_xor(v, 8);
  return v;
}

// A-fragment (16x32 bf16): lane holds row (lane&15); lanes 0-15 get
// K [0,16), lanes 16-31 get [16,32) (contiguous, matches B-fragment).
__device__ __forceinline__ v16bf load_a_frag(const __bf16* rowptr, int kbase, int hl) {
  V16U u;
  u.h[0] = *(const v8bf*)(rowptr + kbase + hl * 16);
  u.h[1] = *(const v8bf*)(rowptr + kbase + hl * 16 + 8);
  return u.v;
}
// B-fragment (32x16 bf16): lane holds col (lane&15), contiguous K run of 16
// starting at (lane>>4)*16. p must point at that run.
__device__ __forceinline__ v16bf load_b_frag(const __bf16* p) {
  V16U u;
  u.h[0] = *(const v8bf*)(p);
  u.h[1] = *(const v8bf*)(p + 8);
  return u.v;
}

__device__ __forceinline__ v8f wmma_bf16(v16bf a, v16bf b, v8f c) {
  return __builtin_amdgcn_wmma_f32_16x16x32_bf16(false, a, false, b, (short)0, c, false, false);
}

// ---- CDNA5 async global->LDS DMA (ASYNCcnt-tracked, no VGPR data path) ----
// vdst = per-lane LDS byte address (low 32 bits of flat shared pointer),
// vaddr = 64-bit global address. 16B per lane.
__device__ __forceinline__ void async_ld128(unsigned lds_off, const void* g) {
  asm volatile("global_load_async_to_lds_b128 %0, %1, off"
               :: "v"(lds_off), "v"((unsigned long long)(uintptr_t)g)
               : "memory");
}
__device__ __forceinline__ void wait_async0() {
  asm volatile("s_wait_asynccnt 0x0" ::: "memory");
}
__device__ __forceinline__ unsigned lds_off32(const void* p) {
  return (unsigned)(uintptr_t)p;   // flat shared addr: low 32 bits = LDS byte addr
}

// ---------------- Kernel 0: fp32 -> bf16 conversion ----------------
__global__ void cvt_kernel(const float* __restrict__ x, const float* __restrict__ wqkv,
                           const float* __restrict__ wproj,
                           __bf16* __restrict__ xb, __bf16* __restrict__ wqkvb,
                           __bf16* __restrict__ wprojb) {
  const long n0 = (long)M_ * C_;
  const long n1 = (long)QKVN_ * C_;
  const long n2 = (long)C_ * C_;
  const long total = n0 + n1 + n2;
  for (long i = blockIdx.x * (long)blockDim.x + threadIdx.x; i < total;
       i += (long)gridDim.x * blockDim.x) {
    if (i < n0)            xb[i]               = (__bf16)x[i];
    else if (i < n0 + n1)  wqkvb[i - n0]       = (__bf16)wqkv[i - n0];
    else                   wprojb[i - n0 - n1] = (__bf16)wproj[i - n0 - n1];
  }
}

// ---------------- LDS-staged block GEMM core (double-buffered, async DMA) ----
// 256 threads = 8 waves compute a TM_ x TN_ tile of A(MxK) * B(NxK)^T with
// K contiguous in both (NT GEMM). Each wave owns a 32x64 register tile
// (waves arranged 4(M) x 2(N)). Fills acc[2][4] (C-fragment layout).
struct GemmRegs { v8f acc[2][4]; };

__device__ __forceinline__ void block_gemm(const __bf16* __restrict__ Ag,
                                           const __bf16* __restrict__ Bg,
                                           int mBase, int nBase, int K,
                                           __bf16* lsA0, __bf16* lsA1,
                                           __bf16* lsB0, __bf16* lsB1, GemmRegs& R) {
  const int t = threadIdx.x;
  const int lane = t & 31, ln = lane & 15, hl = lane >> 4;
  const int wid = t >> 5;
  const int mSub = (wid & 3) * 32;
  const int nSub = (wid >> 2) * 64;

  const v8f vz = {0.f, 0.f, 0.f, 0.f, 0.f, 0.f, 0.f, 0.f};
#pragma unroll
  for (int a = 0; a < 2; ++a)
#pragma unroll
    for (int j = 0; j < 4; ++j) R.acc[a][j] = vz;

  // staging: 512 16B-chunks per tile, 2 per thread (same row)
  const int row0 = t >> 1, q0 = (t & 1) * 2;
  const int ga0 = imin_(mBase + row0, M_ - 1);        // clamp (last M tile)
  const __bf16* agp = Ag + (size_t)ga0 * K + q0 * 8;
  const __bf16* bgp = Bg + (size_t)(nBase + row0) * K + q0 * 8;
  const unsigned aO0 = lds_off32(lsA0 + row0 * LDK_ + q0 * 8);
  const unsigned aO1 = lds_off32(lsA1 + row0 * LDK_ + q0 * 8);
  const unsigned bO0 = lds_off32(lsB0 + row0 * LDK_ + q0 * 8);
  const unsigned bO1 = lds_off32(lsB1 + row0 * LDK_ + q0 * 8);

  auto stage = [&](int k0, unsigned aO, unsigned bO) {
    async_ld128(aO,      agp + k0);
    async_ld128(aO + 16, agp + k0 + 8);
    async_ld128(bO,      bgp + k0);
    async_ld128(bO + 16, bgp + k0 + 8);
  };
  auto compute = [&](const __bf16* cA, const __bf16* cB) {
    v16bf a0 = load_a_frag(cA + (mSub + ln) * LDK_, 0, hl);
    v16bf a1 = load_a_frag(cA + (mSub + 16 + ln) * LDK_, 0, hl);
    v16bf b0 = load_b_frag(cB + (nSub + 0 * 16 + ln) * LDK_ + hl * 16);
    v16bf b1 = load_b_frag(cB + (nSub + 1 * 16 + ln) * LDK_ + hl * 16);
    v16bf b2 = load_b_frag(cB + (nSub + 2 * 16 + ln) * LDK_ + hl * 16);
    v16bf b3 = load_b_frag(cB + (nSub + 3 * 16 + ln) * LDK_ + hl * 16);
    R.acc[0][0] = wmma_bf16(a0, b0, R.acc[0][0]);
    R.acc[1][0] = wmma_bf16(a1, b0, R.acc[1][0]);
    R.acc[0][1] = wmma_bf16(a0, b1, R.acc[0][1]);
    R.acc[1][1] = wmma_bf16(a1, b1, R.acc[1][1]);
    R.acc[0][2] = wmma_bf16(a0, b2, R.acc[0][2]);
    R.acc[1][2] = wmma_bf16(a1, b2, R.acc[1][2]);
    R.acc[0][3] = wmma_bf16(a0, b3, R.acc[0][3]);
    R.acc[1][3] = wmma_bf16(a1, b3, R.acc[1][3]);
  };

  // prologue: DMA tile 0 into buffer 0
  stage(0, aO0, bO0);
  wait_async0();
  __syncthreads();

  for (int k0 = 0; k0 < K; k0 += 2 * TK_) {
    // DMA tile k0+TK_ into buf1 while computing from buf0
    if (k0 + TK_ < K) stage(k0 + TK_, aO1, bO1);
    compute(lsA0, lsB0);
    wait_async0();
    __syncthreads();
    if (k0 + TK_ < K) {
      // DMA tile k0+2*TK_ into buf0 while computing from buf1
      if (k0 + 2 * TK_ < K) stage(k0 + 2 * TK_, aO0, bO0);
      compute(lsA1, lsB1);
      wait_async0();
      __syncthreads();
    }
  }
}

// ---------------- Kernel 1: QKV GEMM (M=18464, N=2304, K=768) ----------------
// Output scattered to Q[b,h,n,d], K[b,h,n,d], V^T[b,h,d,n] (bf16).
__global__ void __launch_bounds__(256)
qkv_gemm_kernel(const __bf16* __restrict__ xb, const __bf16* __restrict__ wb,
                __bf16* __restrict__ qb, __bf16* __restrict__ kb,
                __bf16* __restrict__ vtb) {
  __shared__ __align__(16) __bf16 lsA[2][TM_ * LDK_];
  __shared__ __align__(16) __bf16 lsB[2][TN_ * LDK_];
  const int mB = blockIdx.x % 145;          // ceil(18464/128) = 145
  const int nB = blockIdx.x / 145;          // 2304/128 = 18 (exact)
  const int mBase = mB * TM_, nBase = nB * TN_;

  GemmRegs R;
  block_gemm(xb, wb, mBase, nBase, C_, lsA[0], lsA[1], lsB[0], lsB[1], R);

  const int lane = threadIdx.x & 31, ln = lane & 15, hl = lane >> 4;
  const int wid = threadIdx.x >> 5;
  const int mTile = mBase + (wid & 3) * 32;
  const int nTile = nBase + (wid >> 2) * 64;

#pragma unroll
  for (int mi = 0; mi < 2; ++mi) {
#pragma unroll
    for (int r = 0; r < 8; ++r) {
      const int m = mTile + mi * 16 + r + 8 * hl;
      if (m < M_) {
        const int bb = m / 577, nn = m % 577;
#pragma unroll
        for (int j = 0; j < 4; ++j) {
          const int col = nTile + j * 16;           // s,h uniform within frag
          const int s = col / 768;
          const int rem = col - s * 768;
          const int hh = rem >> 6;
          const int d = (rem & 63) + ln;
          const __bf16 val = (__bf16)R.acc[mi][j][r];
          if (s == 0)
            qb[(((size_t)bb * H_ + hh) * N_ + nn) * HD_ + d] = val;
          else if (s == 1)
            kb[(((size_t)bb * H_ + hh) * N_ + nn) * HD_ + d] = val;
          else
            vtb[(((size_t)bb * H_ + hh) * HD_ + d) * VSTRIDE_ + nn] = val;
        }
      }
    }
  }
}

// ---------------- Kernel 2: flash attention ----------------
// One wave: 16 query rows x full HD=64, streaming softmax over key tiles of 32.
__global__ void attn_kernel(const __bf16* __restrict__ qb, const __bf16* __restrict__ kb,
                            const __bf16* __restrict__ vtb, const float* __restrict__ mask,
                            __bf16* __restrict__ obnc) {
  __shared__ __align__(16) __bf16 plds[4 * 16 * 32];   // 1KB per wave, wave-private
  const int wib = threadIdx.x >> 5;
  const int wave = blockIdx.x * (blockDim.x >> 5) + wib;
  const int lane = threadIdx.x & 31;
  const int ln = lane & 15, hl = lane >> 4;
  const int qt = wave % 37;                 // 37 q-tiles of 16 (577 -> 592)
  const int h = (wave / 37) % H_;
  const int b = wave / (37 * H_);
  const int qbase = qt * 16;
  __bf16* pl = plds + wib * 512;

  const __bf16* qhb = qb + ((size_t)b * H_ + h) * N_ * HD_;
  const __bf16* khb = kb + ((size_t)b * H_ + h) * N_ * HD_;
  const __bf16* vhb = vtb + ((size_t)b * H_ + h) * HD_ * VSTRIDE_;
  const float* mh = mask + (size_t)h * N_ * N_;

  // Q fragments, kept in registers for the whole kernel (A-frag: row = ln)
  const int qr = imin_(qbase + ln, N_ - 1);
  const v16bf qa0 = load_a_frag(qhb + (size_t)qr * HD_, 0, hl);    // d 0..31
  const v16bf qa1 = load_a_frag(qhb + (size_t)qr * HD_, 32, hl);   // d 32..63

  const v8f vzero = {0.f, 0.f, 0.f, 0.f, 0.f, 0.f, 0.f, 0.f};
  v8f acc[4];
#pragma unroll
  for (int j = 0; j < 4; ++j) acc[j] = vzero;
  float mrow[8], srow[8];
#pragma unroll
  for (int r = 0; r < 8; ++r) { mrow[r] = -3.0e38f; srow[r] = 0.f; }

  for (int kt = 0; kt < N_; kt += 32) {
    // S = Q K^T for this 16q x 32k tile (two 16x16 C-frags)
    v8f s0 = vzero, s1 = vzero;
    {
      const int key = imin_(kt + ln, N_ - 1);
      const __bf16* kp = khb + (size_t)key * HD_;
      s0 = wmma_bf16(qa0, load_b_frag(kp + hl * 16), s0);
      s0 = wmma_bf16(qa1, load_b_frag(kp + 32 + hl * 16), s0);
    }
    {
      const int key = imin_(kt + 16 + ln, N_ - 1);
      const __bf16* kp = khb + (size_t)key * HD_;
      s1 = wmma_bf16(qa0, load_b_frag(kp + hl * 16), s1);
      s1 = wmma_bf16(qa1, load_b_frag(kp + 32 + hl * 16), s1);
    }
    // scale + additive mask + OOB masking + online softmax update
    const int k0c = kt + ln, k1c = kt + 16 + ln;
    const int k0cl = imin_(k0c, N_ - 1), k1cl = imin_(k1c, N_ - 1);
#pragma unroll
    for (int r = 0; r < 8; ++r) {
      const int qrow = qbase + r + 8 * hl;
      const int qc = imin_(qrow, N_ - 1);
      const float m0 = mh[(size_t)qc * N_ + k0cl];
      const float m1 = mh[(size_t)qc * N_ + k1cl];
      float v0 = (k0c < N_) ? (s0[r] * SCALE_ + m0) : -1.0e30f;
      float v1 = (k1c < N_) ? (s1[r] * SCALE_ + m1) : -1.0e30f;
      const float tmax = fmaxf(rowmax16(v0), rowmax16(v1));
      const float mnew = fmaxf(mrow[r], tmax);
      const float corr = __expf(mrow[r] - mnew);
      mrow[r] = mnew;
      srow[r] *= corr;
#pragma unroll
      for (int j = 0; j < 4; ++j) acc[j][r] *= corr;  // correction in C-layout
      const float e0 = __expf(v0 - mnew);
      const float e1 = __expf(v1 - mnew);
      srow[r] += rowsum16(e0 + e1);
      const int qrel = r + 8 * hl;
      pl[qrel * 32 + ln] = (__bf16)e0;        // P tile to LDS, [16q][32k]
      pl[qrel * 32 + 16 + ln] = (__bf16)e1;
    }
    // reload P in A-fragment layout (same-wave DS ops are in-order)
    v16bf pa;
    {
      V16U u;
      u.h[0] = *(const v8bf*)(pl + ln * 32 + hl * 16);
      u.h[1] = *(const v8bf*)(pl + ln * 32 + hl * 16 + 8);
      pa = u.v;
    }
    // O += P @ V  (V^T layout -> contiguous B-frag loads; OOB keys have P==0)
#pragma unroll
    for (int j = 0; j < 4; ++j) {
      v16bf bv = load_b_frag(vhb + (size_t)(j * 16 + ln) * VSTRIDE_ + kt + hl * 16);
      acc[j] = wmma_bf16(pa, bv, acc[j]);
    }
  }

  // epilogue: normalize and store O in [B,N,C] bf16 for the proj GEMM
#pragma unroll
  for (int r = 0; r < 8; ++r) {
    const int qrow = qbase + r + 8 * hl;
    if (qrow < N_) {
      const float inv = 1.0f / srow[r];
#pragma unroll
      for (int j = 0; j < 4; ++j)
        obnc[((size_t)b * N_ + qrow) * C_ + h * HD_ + j * 16 + ln] = (__bf16)(acc[j][r] * inv);
    }
  }
}

// ---------------- Kernel 3: proj GEMM (M=18464, N=768, K=768) + bias ----------------
__global__ void __launch_bounds__(256)
proj_gemm_kernel(const __bf16* __restrict__ ob, const __bf16* __restrict__ wb,
                 const float* __restrict__ bias, float* __restrict__ out) {
  __shared__ __align__(16) __bf16 lsA[2][TM_ * LDK_];
  __shared__ __align__(16) __bf16 lsB[2][TN_ * LDK_];
  const int mB = blockIdx.x % 145;
  const int nB = blockIdx.x / 145;          // 768/128 = 6 (exact)
  const int mBase = mB * TM_, nBase = nB * TN_;

  GemmRegs R;
  block_gemm(ob, wb, mBase, nBase, C_, lsA[0], lsA[1], lsB[0], lsB[1], R);

  const int lane = threadIdx.x & 31, ln = lane & 15, hl = lane >> 4;
  const int wid = threadIdx.x >> 5;
  const int mTile = mBase + (wid & 3) * 32;
  const int nTile = nBase + (wid >> 2) * 64;

#pragma unroll
  for (int mi = 0; mi < 2; ++mi) {
#pragma unroll
    for (int r = 0; r < 8; ++r) {
      const int m = mTile + mi * 16 + r + 8 * hl;
      if (m < M_) {
#pragma unroll
        for (int j = 0; j < 4; ++j) {
          const int col = nTile + j * 16 + ln;
          out[(size_t)m * C_ + col] = R.acc[mi][j][r] + bias[col];
        }
      }
    }
  }
}

extern "C" void kernel_launch(void* const* d_in, const int* in_sizes, int n_in,
                              void* d_out, int out_size, void* d_ws, size_t ws_size,
                              hipStream_t stream) {
  const float* x      = (const float*)d_in[0];
  const float* qkv_w  = (const float*)d_in[1];
  const float* proj_w = (const float*)d_in[2];
  const float* proj_b = (const float*)d_in[3];
  const float* attn_m = (const float*)d_in[4];
  float* out = (float*)d_out;

  char* ws = (char*)d_ws;
  size_t off = 0;
  auto wsalloc = [&](size_t bytes) -> void* {
    void* p = ws + off;
    off += (bytes + 255) & ~(size_t)255;
    return p;
  };
  __bf16* xb     = (__bf16*)wsalloc((size_t)M_ * C_ * 2);
  __bf16* wqkvb  = (__bf16*)wsalloc((size_t)QKVN_ * C_ * 2);
  __bf16* wprojb = (__bf16*)wsalloc((size_t)C_ * C_ * 2);
  __bf16* qb     = (__bf16*)wsalloc((size_t)B_ * H_ * N_ * HD_ * 2);
  __bf16* kb     = (__bf16*)wsalloc((size_t)B_ * H_ * N_ * HD_ * 2);
  __bf16* vtb    = (__bf16*)wsalloc((size_t)B_ * H_ * HD_ * VSTRIDE_ * 2);
  __bf16* obnc   = (__bf16*)wsalloc((size_t)M_ * C_ * 2);

  cvt_kernel<<<2048, 256, 0, stream>>>(x, qkv_w, proj_w, xb, wqkvb, wprojb);
  // 145 M-tiles x 18 N-tiles of 128x128
  qkv_gemm_kernel<<<145 * 18, 256, 0, stream>>>(xb, wqkvb, qb, kb, vtb);
  // 32 * 12 * 37 = 14208 waves -> 3552 blocks of 4 waves
  attn_kernel<<<3552, 128, 0, stream>>>(qb, kb, vtb, attn_m, obnc);
  // 145 M-tiles x 6 N-tiles
  proj_gemm_kernel<<<145 * 6, 256, 0, stream>>>(obnc, wprojb, proj_b, out);
}
